// HyperER_47605417509202
// MI455X (gfx1250) — matrix operand order repaired and checked by reading, not compiled, not run on hardware
//
#include <hip/hip_runtime.h>

// Problem constants (from reference)
#define ED    200
#define KW    9
#define OC    32
#define WOUT  192              // ED - KW + 1
#define FEAT  (OC * WOUT)      // 6144
#define BATCH 1024
#define NE    40943
#define NR    18
#define KWOC  (KW * OC)        // 288
#define EDP   224              // ED padded to multiple of 32 for K-loop

typedef __attribute__((ext_vector_type(8)))  float  v8f;
typedef __attribute__((ext_vector_type(8)))  __bf16 v8bf;
typedef __attribute__((ext_vector_type(16))) __bf16 v16bf;

__device__ __forceinline__ void split_bf16(float x, __bf16& hi, __bf16& lo) {
    hi = (__bf16)x;
    lo = (__bf16)(x - (float)hi);
}

// A-operand (16x32 bf16): lane holds K chunks [k0+8*hl, +8) and [k0+16+8*hl, +8)
__device__ __forceinline__ v16bf load_Afrag(const __bf16* __restrict__ row, int k0, int hl) {
    v8bf c0 = *(const v8bf*)(row + k0 + 8 * hl);
    v8bf c1 = *(const v8bf*)(row + k0 + 16 + 8 * hl);
    return __builtin_shufflevector(c0, c1, 0,1,2,3,4,5,6,7,8,9,10,11,12,13,14,15);
}
// B-operand (32x16 bf16): lane holds K chunks [k0+8*hl, +8) and [k0+16+8*hl, +8)
__device__ __forceinline__ v16bf load_Bfrag(const __bf16* __restrict__ row, int k0, int hl) {
    v8bf c0 = *(const v8bf*)(row + k0 + 8 * hl);
    v8bf c1 = *(const v8bf*)(row + k0 + 16 + 8 * hl);
    return __builtin_shufflevector(c0, c1, 0,1,2,3,4,5,6,7,8,9,10,11,12,13,14,15);
}

#define WMMA_BF16(A, B, C) __builtin_amdgcn_wmma_f32_16x16x32_bf16(false, (A), false, (B), (short)0, (C), false, false)

// ---------------------------------------------------------------------------
// Kernel 1: Kmat[r] = b1 + R[r] @ W1   (18 x 288, trivial, f32)
// ---------------------------------------------------------------------------
__global__ __launch_bounds__(KWOC)
void hyper_k_precompute(const float* __restrict__ R, const float* __restrict__ W1,
                        const float* __restrict__ b1, float* __restrict__ Kmat) {
    const int r = blockIdx.x, col = threadIdx.x;
    const float* Rr = R + r * ED;
    float acc = b1[col];
    #pragma unroll 4
    for (int i = 0; i < ED; ++i)
        acc = fmaf(Rr[i], W1[i * KWOC + col], acc);
    Kmat[r * KWOC + col] = acc;
}

// ---------------------------------------------------------------------------
// Kernel 2: E (NE x 200 f32) -> E_hi/E_lo (NE x 224 bf16, zero-padded K)
// ---------------------------------------------------------------------------
__global__ __launch_bounds__(256)
void hyper_convert_E(const float* __restrict__ E, __bf16* __restrict__ E_hi,
                     __bf16* __restrict__ E_lo) {
    const int total = NE * EDP;
    for (int idx = blockIdx.x * 256 + threadIdx.x; idx < total; idx += gridDim.x * 256) {
        const int row = idx / EDP, c = idx % EDP;
        const float x = (c < ED) ? E[row * ED + c] : 0.f;
        __bf16 hi, lo; split_bf16(x, hi, lo);
        E_hi[idx] = hi; E_lo[idx] = lo;
    }
}

// ---------------------------------------------------------------------------
// Kernel 3: W2 (6144 x 200 f32) -> W2t_hi/lo (200 x 6144 bf16), LDS transpose
// ---------------------------------------------------------------------------
__global__ __launch_bounds__(256)
void hyper_transpose_W2(const float* __restrict__ W2, __bf16* __restrict__ Thi,
                        __bf16* __restrict__ Tlo) {
    __shared__ float tile[32][33];
    const int tx = threadIdx.x, ty = threadIdx.y;          // 32 x 8
    const int i0 = blockIdx.x * 32, d0 = blockIdx.y * 32;
    #pragma unroll
    for (int k = 0; k < 4; ++k) {
        const int i = i0 + ty + 8 * k, d = d0 + tx;
        tile[ty + 8 * k][tx] = (d < ED) ? W2[i * ED + d] : 0.f;
    }
    __syncthreads();
    #pragma unroll
    for (int k = 0; k < 4; ++k) {
        const int d = d0 + ty + 8 * k, i = i0 + tx;
        if (d < ED) {
            __bf16 hi, lo; split_bf16(tile[tx][ty + 8 * k], hi, lo);
            Thi[d * FEAT + i] = hi; Tlo[d * FEAT + i] = lo;
        }
    }
}

// ---------------------------------------------------------------------------
// Kernel 4: zero h_hi/h_lo (covers the K-pad region 200..223)
// ---------------------------------------------------------------------------
__global__ __launch_bounds__(256)
void hyper_zero_h(__bf16* __restrict__ h_hi, __bf16* __restrict__ h_lo) {
    const int idx = blockIdx.x * 256 + threadIdx.x;   // grid sized exactly
    h_hi[idx] = (__bf16)0.f; h_lo[idx] = (__bf16)0.f;
}

// ---------------------------------------------------------------------------
// Kernel 5: sliding-window einsum -> feat_hi/feat_lo (bf16), 2 samples/block
// ---------------------------------------------------------------------------
__global__ __launch_bounds__(256)
void hyper_feat(const int* __restrict__ e1_idx, const int* __restrict__ r_idx,
                const float* __restrict__ E, const float* __restrict__ Kmat,
                __bf16* __restrict__ feat_hi, __bf16* __restrict__ feat_lo) {
    __shared__ float e1s[2][ED + 8];
    __shared__ float ks[2][KWOC];
    const int tid = threadIdx.x, b0 = blockIdx.x * 2;
    for (int bb = 0; bb < 2; ++bb) {
        const float* erow = E + (long)e1_idx[b0 + bb] * ED;
        for (int i = tid; i < ED; i += 256) e1s[bb][i] = erow[i];
        const float* krow = Kmat + r_idx[b0 + bb] * KWOC;
        for (int i = tid; i < KWOC; i += 256) ks[bb][i] = krow[i];
    }
    __syncthreads();
    for (int idx = tid; idx < 2 * FEAT; idx += 256) {
        const int bb = idx / FEAT, rem = idx % FEAT;
        const int o = rem / WOUT, w = rem % WOUT;
        float acc = 0.f;
        #pragma unroll
        for (int j = 0; j < KW; ++j)
            acc = fmaf(e1s[bb][w + j], ks[bb][j * OC + o], acc);
        __bf16 hi, lo; split_bf16(acc, hi, lo);
        feat_hi[(b0 + bb) * FEAT + rem] = hi;
        feat_lo[(b0 + bb) * FEAT + rem] = lo;
    }
}

// ---------------------------------------------------------------------------
// Kernel 6: h = relu(feat @ W2t^T + b2) via split-bf16 WMMA
//   M=1024, N=200 (13 tiles), K=6144 (192 steps). 1 tile per wave.
// ---------------------------------------------------------------------------
#define NT3 ((ED + 15) / 16)     // 13
__global__ __launch_bounds__(256)
void hyper_h_wmma(const __bf16* __restrict__ feat_hi, const __bf16* __restrict__ feat_lo,
                  const __bf16* __restrict__ W2t_hi, const __bf16* __restrict__ W2t_lo,
                  const float* __restrict__ b2,
                  __bf16* __restrict__ h_hi, __bf16* __restrict__ h_lo) {
    const int wave = threadIdx.x >> 5, lane = threadIdx.x & 31;
    const int gw = blockIdx.x * 8 + wave;
    if (gw >= (BATCH / 16) * NT3) return;
    const int mt = gw / NT3, ntile = gw % NT3;
    const int M0 = mt * 16, N0 = ntile * 16;
    const int rsel = lane & 15, hl = lane >> 4;

    const __bf16* Ahi = feat_hi + (M0 + rsel) * FEAT;
    const __bf16* Alo = feat_lo + (M0 + rsel) * FEAT;
    int nn = N0 + rsel; nn = (nn < ED) ? nn : (ED - 1);
    const __bf16* Bhi = W2t_hi + nn * FEAT;
    const __bf16* Blo = W2t_lo + nn * FEAT;
    __builtin_prefetch(Bhi, 0, 0);
    __builtin_prefetch(Blo, 0, 0);

    v8f acc = (v8f){0.f,0.f,0.f,0.f,0.f,0.f,0.f,0.f};
    for (int k0 = 0; k0 < FEAT; k0 += 32) {
        const v16bf ah = load_Afrag(Ahi, k0, hl);
        const v16bf al = load_Afrag(Alo, k0, hl);
        const v16bf bh = load_Bfrag(Bhi, k0, hl);
        const v16bf bl = load_Bfrag(Blo, k0, hl);
        acc = WMMA_BF16(ah, bh, acc);
        acc = WMMA_BF16(ah, bl, acc);
        acc = WMMA_BF16(al, bh, acc);
    }
    const int n = N0 + rsel;
    if (n < ED) {
        const float bv = b2[n];
        #pragma unroll
        for (int v = 0; v < 8; ++v) {
            const int m = M0 + v + 8 * hl;
            const float r = fmaxf(acc[v] + bv, 0.f);
            __bf16 hi, lo; split_bf16(r, hi, lo);
            h_hi[m * EDP + n] = hi; h_lo[m * EDP + n] = lo;
        }
    }
}

// ---------------------------------------------------------------------------
// Kernel 7: logits = h @ E^T + bias via split-bf16 WMMA
//   M=1024 (64 tiles), N=40943 (2559 tiles), K=224 (7 steps).
//   Each wave: 1 M-tile x 4 N-tiles.
// ---------------------------------------------------------------------------
#define NT 4
#define NTILES  ((NE + 15) / 16)            // 2559
#define NGROUPS ((NTILES + NT - 1) / NT)    // 640
#define MTILES  (BATCH / 16)                // 64
__global__ __launch_bounds__(256)
void hyper_logits_wmma(const __bf16* __restrict__ h_hi, const __bf16* __restrict__ h_lo,
                       const __bf16* __restrict__ E_hi, const __bf16* __restrict__ E_lo,
                       const float* __restrict__ bias, float* __restrict__ out) {
    const int wave = threadIdx.x >> 5, lane = threadIdx.x & 31;
    const int gw = blockIdx.x * 8 + wave;
    const int mt = gw / NGROUPS, ng = gw % NGROUPS;
    if (mt >= MTILES) return;
    const int M0 = mt * 16, N0 = ng * (NT * 16);
    const int rsel = lane & 15, hl = lane >> 4;

    const __bf16* Ahi = h_hi + (M0 + rsel) * EDP;
    const __bf16* Alo = h_lo + (M0 + rsel) * EDP;
    const __bf16* Bhi[NT];
    const __bf16* Blo[NT];
    #pragma unroll
    for (int nt = 0; nt < NT; ++nt) {
        int nn = N0 + nt * 16 + rsel;
        nn = (nn < NE) ? nn : (NE - 1);
        Bhi[nt] = E_hi + (long)nn * EDP;
        Blo[nt] = E_lo + (long)nn * EDP;
        __builtin_prefetch(Bhi[nt], 0, 0);
        __builtin_prefetch(Blo[nt], 0, 0);
    }

    v8f acc[NT];
    #pragma unroll
    for (int nt = 0; nt < NT; ++nt) acc[nt] = (v8f){0.f,0.f,0.f,0.f,0.f,0.f,0.f,0.f};

    #pragma unroll
    for (int k0 = 0; k0 < EDP; k0 += 32) {     // 7 steps
        const v16bf ah = load_Afrag(Ahi, k0, hl);
        const v16bf al = load_Afrag(Alo, k0, hl);
        #pragma unroll
        for (int nt = 0; nt < NT; ++nt) {
            const v16bf bh = load_Bfrag(Bhi[nt], k0, hl);
            const v16bf bl = load_Bfrag(Blo[nt], k0, hl);
            acc[nt] = WMMA_BF16(ah, bh, acc[nt]);
            acc[nt] = WMMA_BF16(ah, bl, acc[nt]);
            acc[nt] = WMMA_BF16(al, bh, acc[nt]);
        }
    }

    #pragma unroll
    for (int nt = 0; nt < NT; ++nt) {
        const int n = N0 + nt * 16 + rsel;
        if (n < NE) {
            const float bv = bias[n];
            #pragma unroll
            for (int v = 0; v < 8; ++v) {
                const int m = M0 + v + 8 * hl;
                out[(long)m * NE + n] = acc[nt][v] + bv;
            }
        }
    }
}

// ---------------------------------------------------------------------------
// Host-side launcher
// ---------------------------------------------------------------------------
extern "C" void kernel_launch(void* const* d_in, const int* in_sizes, int n_in,
                              void* d_out, int out_size, void* d_ws, size_t ws_size,
                              hipStream_t stream) {
    const int*   e1_idx = (const int*)  d_in[0];
    const int*   r_idx  = (const int*)  d_in[1];
    const float* W1     = (const float*)d_in[2];
    const float* b1     = (const float*)d_in[3];
    const float* W2     = (const float*)d_in[4];
    const float* b2     = (const float*)d_in[5];
    const float* bias   = (const float*)d_in[6];
    const float* E      = (const float*)d_in[7];
    const float* R      = (const float*)d_in[8];
    float* out = (float*)d_out;

    // Workspace layout (256B aligned slabs)
    char* base = (char*)d_ws;
    size_t off = 0;
    auto take = [&](size_t bytes) { char* p = base + off; off = (off + bytes + 255) & ~(size_t)255; return p; };
    float*  Kmat    = (float*) take(NR * KWOC * sizeof(float));
    __bf16* E_hi    = (__bf16*)take((size_t)NE * EDP * 2);
    __bf16* E_lo    = (__bf16*)take((size_t)NE * EDP * 2);
    __bf16* feat_hi = (__bf16*)take((size_t)BATCH * FEAT * 2);
    __bf16* feat_lo = (__bf16*)take((size_t)BATCH * FEAT * 2);
    __bf16* W2t_hi  = (__bf16*)take((size_t)ED * FEAT * 2);
    __bf16* W2t_lo  = (__bf16*)take((size_t)ED * FEAT * 2);
    __bf16* h_hi    = (__bf16*)take((size_t)BATCH * EDP * 2);
    __bf16* h_lo    = (__bf16*)take((size_t)BATCH * EDP * 2);

    hyper_k_precompute<<<NR, KWOC, 0, stream>>>(R, W1, b1, Kmat);
    hyper_convert_E<<<8192, 256, 0, stream>>>(E, E_hi, E_lo);
    hyper_transpose_W2<<<dim3(FEAT / 32, (ED + 31) / 32), dim3(32, 8), 0, stream>>>(W2, W2t_hi, W2t_lo);
    hyper_zero_h<<<(BATCH * EDP) / 256, 256, 0, stream>>>(h_hi, h_lo);
    hyper_feat<<<BATCH / 2, 256, 0, stream>>>(e1_idx, r_idx, E, Kmat, feat_hi, feat_lo);

    const int waves3 = (BATCH / 16) * NT3;                 // 832
    hyper_h_wmma<<<(waves3 + 7) / 8, 256, 0, stream>>>(feat_hi, feat_lo, W2t_hi, W2t_lo, b2, h_hi, h_lo);

    const int waves4 = MTILES * NGROUPS;                   // 40960
    hyper_logits_wmma<<<waves4 / 8, 256, 0, stream>>>(h_hi, h_lo, E_hi, E_lo, bias, out);
}
